// Model_2869038154100
// MI455X (gfx1250) — matrix operand order repaired and checked by reading, not compiled, not run on hardware
//
#include <hip/hip_runtime.h>

// ---------------------------------------------------------------------------
// CDNA5 (gfx1250) WMMA GGNN forward pass, round 2.
// Node state h stored in bf16 (halves the dominant gather stream); all GEMMs
// on v_wmma_f32_16x16x32_bf16 with f32 accumulation; scatter-add in f32.
// ---------------------------------------------------------------------------

typedef __attribute__((ext_vector_type(16))) __bf16 v16bf;
typedef __attribute__((ext_vector_type(8)))  __bf16 v8bf;
typedef __attribute__((ext_vector_type(8)))  float  v8f;

#define NN   100000
#define EE   800000
#define BB   4096
#define HH   128
#define DD   300
#define DDP  320
#define LL   2
#define TT   3

__device__ __forceinline__ v8f wmma_bf16(v16bf a, v16bf b, v8f c) {
  return __builtin_amdgcn_wmma_f32_16x16x32_bf16(
      /*neg_a=*/false, a, /*neg_b=*/false, b,
      /*c_mod=*/(short)0, c, /*reuse_a=*/false, /*reuse_b=*/false);
}

// A-fragment (16x32 bf16, MxK): lane L holds row M=L&15.
// lanes 0-15 : elems 0-15 => K=k+0..15
// lanes 16-31: elems 0-15 => K=k+16..31
// f32 source: bounds-checked scalar loads + cvt.
template <bool BOUNDED>
__device__ __forceinline__ v16bf make_a_frag_f32(const float* __restrict__ rowptr,
                                                 int k, int K) {
  const int lane = threadIdx.x & 31;
  const int kb = (lane & 16) ? 16 : 0;
  v16bf a;
#pragma unroll
  for (int e = 0; e < 8; ++e) {
    int kk = k + kb + e;
    float v = (!BOUNDED || kk < K) ? rowptr[kk] : 0.0f;
    a[e] = (__bf16)v;
  }
#pragma unroll
  for (int e = 0; e < 8; ++e) {
    int kk = k + kb + 8 + e;
    float v = (!BOUNDED || kk < K) ? rowptr[kk] : 0.0f;
    a[8 + e] = (__bf16)v;
  }
  return a;
}

// bf16 source: two aligned 16-byte loads, no conversion.
__device__ __forceinline__ v16bf make_a_frag_bf16(const __bf16* __restrict__ rowptr,
                                                  int k) {
  const int lane = threadIdx.x & 31;
  const int kb = (lane & 16) ? 16 : 0;
  v8bf lo = *(const v8bf*)(rowptr + k + kb);
  v8bf hi = *(const v8bf*)(rowptr + k + kb + 8);
  v16bf a;
#pragma unroll
  for (int e = 0; e < 8; ++e) { a[e] = lo[e]; a[8 + e] = hi[e]; }
  return a;
}

// B-fragment (32x16 bf16, KxN) from row-major W[Hout x Kpad] (B = W^T).
// lane L holds column N=L&15; lanes 0-15 K=k+0..15, lanes 16-31 K=k+16..31.
__device__ __forceinline__ v16bf make_b_frag(const __bf16* __restrict__ W,
                                             int Kpad, int n0, int k) {
  const int lane = threadIdx.x & 31;
  const int n  = n0 + (lane & 15);
  const int kh = (lane & 16) ? 16 : 0;
  return *(const v16bf*)(W + (size_t)n * Kpad + k + kh);
}

__device__ __forceinline__ float sigm(float x) {
  return 1.0f / (1.0f + __expf(-x));
}
__device__ __forceinline__ float tanh_fast(float x) {
  float cx = fminf(fmaxf(x, -15.0f), 15.0f);
  float e = __expf(2.0f * cx);
  return (e - 1.0f) / (e + 1.0f);
}

// ---------------------------------------------------------------------------
// f32 -> bf16 weight conversion with K padding (pad region zeroed).
// ---------------------------------------------------------------------------
__global__ void cvt_bf16_kernel(const float* __restrict__ src,
                                __bf16* __restrict__ dst,
                                int rows, int K, int Kpad) {
  int i = blockIdx.x * blockDim.x + threadIdx.x;
  if (i >= rows * Kpad) return;
  int r = i / Kpad, k = i - r * Kpad;
  float v = (k < K) ? src[(size_t)r * K + k] : 0.0f;
  dst[i] = (__bf16)v;
}

// ---------------------------------------------------------------------------
// Generic gathered GEMM: out[i,n] = act( bf16(X[idx[i],:K]) @ W^T + bias )
// One wave = one 16x16 output tile; wave-uniform guard => EXEC all-ones.
// OT = float or __bf16 output.
// ---------------------------------------------------------------------------
template <bool BOUNDED, typename OT>
__global__ void gather_gemm_kernel(const float* __restrict__ X, int ldx, int K,
                                   int ksteps, const int* __restrict__ idx,
                                   const __bf16* __restrict__ W, int Kpad,
                                   const float* __restrict__ bias,
                                   OT* __restrict__ out, int ldo,
                                   int rowTiles, int colTiles, int act) {
  const int tile = blockIdx.x * (blockDim.x >> 5) + (threadIdx.x >> 5);
  if (tile >= rowTiles * colTiles) return;
  const int rt = tile / colTiles, ct = tile - rt * colTiles;
  const int lane = threadIdx.x & 31;
  const int rowbase = rt * 16, n0 = ct * 16;

  const int m = rowbase + (lane & 15);
  const int r = idx ? idx[m] : m;
  const float* rowptr = X + (size_t)r * ldx;

  v8f acc = {};
  for (int kc = 0; kc < ksteps; ++kc) {
    const int k = kc * 32;
    v16bf a = make_a_frag_f32<BOUNDED>(rowptr, k, K);
    v16bf b = make_b_frag(W, Kpad, n0, k);
    acc = wmma_bf16(a, b, acc);
  }

  const int n = n0 + (lane & 15);
  const int roff = (lane & 16) ? 8 : 0;
  const float bv = bias ? bias[n] : 0.0f;
#pragma unroll
  for (int rr = 0; rr < 8; ++rr) {
    float v = acc[rr] + bv;
    if (act == 1) v = fmaxf(v, 0.0f);
    out[(size_t)(rowbase + rr + roff) * ldo + n] = (OT)v;
  }
}

// ---------------------------------------------------------------------------
// Fused message GEMM + scatter-add:
//   inc[tgt[e], :] += bf16 h[src[e], :] @ msg_W^T + msg_b   (f32 atomics)
// msg_W staged in LDS (padded rows to dodge bank conflicts).
// One wave = 16 edges x 128 cols (8 col-tiles, 4 K-chunks => 32 WMMAs).
// ---------------------------------------------------------------------------
#define LDSROW 144  // 128 bf16 + 16 bf16 pad (row stride 288 B, 32B aligned)

__global__ void msg_scatter_kernel(const __bf16* __restrict__ h,
                                   const int* __restrict__ adj,
                                   const __bf16* __restrict__ Wg,
                                   const float* __restrict__ bias,
                                   float* __restrict__ inc, int edgeTiles) {
  __shared__ __align__(32) __bf16 Wlds[128 * LDSROW];
  for (int i = threadIdx.x; i < 128 * 8; i += blockDim.x) {
    int r = i >> 3, c = i & 7;
    *(uint4*)(Wlds + r * LDSROW + c * 8) = *(const uint4*)(Wg + r * 128 + c * 8);
  }
  __syncthreads();

  const int et = blockIdx.x * (blockDim.x >> 5) + (threadIdx.x >> 5);
  if (et >= edgeTiles) return;
  const int lane = threadIdx.x & 31;
  const int m = et * 16 + (lane & 15);
  const int src = adj[2 * (size_t)m + 0];
  const __bf16* rowptr = h + (size_t)src * HH;

  v8f acc[8];
#pragma unroll
  for (int ct = 0; ct < 8; ++ct) acc[ct] = (v8f){};

#pragma unroll
  for (int kc = 0; kc < 4; ++kc) {
    const int k = kc * 32;
    v16bf a = make_a_frag_bf16(rowptr, k);
#pragma unroll
    for (int ct = 0; ct < 8; ++ct) {
      const int n = ct * 16 + (lane & 15);
      const int kh = (lane & 16) ? 16 : 0;
      v16bf b = *(const v16bf*)(Wlds + (size_t)n * LDSROW + k + kh);
      acc[ct] = wmma_bf16(a, b, acc[ct]);
    }
  }

  const int nlo = lane & 15;
  const int roff = (lane & 16) ? 8 : 0;
  float bv[8];
#pragma unroll
  for (int ct = 0; ct < 8; ++ct) bv[ct] = bias[ct * 16 + nlo];

#pragma unroll
  for (int rr = 0; rr < 8; ++rr) {
    const int erow = et * 16 + rr + roff;
    const int tgt = adj[2 * (size_t)erow + 1];
    float* dstrow = inc + (size_t)tgt * HH;
#pragma unroll
    for (int ct = 0; ct < 8; ++ct) {
      atomicAdd(dstrow + ct * 16 + nlo, acc[ct][rr] + bv[ct]);
    }
  }
}

// ---------------------------------------------------------------------------
// Fully fused GRU cell: 6 WMMA accumulators (r/z/n x {x@Wih^T, h@Whh^T}),
// gates in-register; x (incoming) is f32, h is bf16, output h is bf16.
// ---------------------------------------------------------------------------
__global__ void gru_kernel(const float* __restrict__ x,
                           const __bf16* __restrict__ h,
                           const __bf16* __restrict__ Wih,
                           const __bf16* __restrict__ Whh,
                           const float* __restrict__ bih,
                           const float* __restrict__ bhh,
                           __bf16* __restrict__ hout, int rowTiles) {
  const int tile = blockIdx.x * (blockDim.x >> 5) + (threadIdx.x >> 5);
  if (tile >= rowTiles * 8) return;
  const int rt = tile >> 3, ct = tile & 7;
  const int lane = threadIdx.x & 31;
  const int rowbase = rt * 16, n0 = ct * 16;

  const int m = rowbase + (lane & 15);
  const float*  xrow = x + (size_t)m * HH;
  const __bf16* hrow = h + (size_t)m * HH;

  v8f gi[3], gh[3];
#pragma unroll
  for (int g = 0; g < 3; ++g) { gi[g] = (v8f){}; gh[g] = (v8f){}; }

#pragma unroll
  for (int kc = 0; kc < 4; ++kc) {
    const int k = kc * 32;
    v16bf ax = make_a_frag_f32<false>(xrow, k, HH);
    v16bf ah = make_a_frag_bf16(hrow, k);
#pragma unroll
    for (int g = 0; g < 3; ++g) {
      v16bf bi = make_b_frag(Wih, HH, n0 + g * HH, k);
      gi[g] = wmma_bf16(ax, bi, gi[g]);
      v16bf bh = make_b_frag(Whh, HH, n0 + g * HH, k);
      gh[g] = wmma_bf16(ah, bh, gh[g]);
    }
  }

  const int n = n0 + (lane & 15);
  const int roff = (lane & 16) ? 8 : 0;
  const float bir = bih[n], biz = bih[n + HH], bin = bih[n + 2 * HH];
  const float bhr = bhh[n], bhz = bhh[n + HH], bhn = bhh[n + 2 * HH];
#pragma unroll
  for (int rr = 0; rr < 8; ++rr) {
    const int row = rowbase + rr + roff;
    float ir = gi[0][rr] + bir, iz = gi[1][rr] + biz, in = gi[2][rr] + bin;
    float hr = gh[0][rr] + bhr, hz = gh[1][rr] + bhz, hn = gh[2][rr] + bhn;
    float rg = sigm(ir + hr);
    float zg = sigm(iz + hz);
    float ng = tanh_fast(in + rg * hn);
    float hv = (float)h[(size_t)row * HH + n];
    hout[(size_t)row * HH + n] = (__bf16)((1.0f - zg) * ng + zg * hv);
  }
}

// ---------------------------------------------------------------------------
// Classifier helpers
// ---------------------------------------------------------------------------
__global__ void prop_gather_kernel(const __bf16* __restrict__ h,
                                   const int* __restrict__ ind,
                                   float* __restrict__ ge_cat, int colOff) {
  int i = blockIdx.x * blockDim.x + threadIdx.x;
  if (i >= BB * HH) return;
  int b = i >> 7, c = i & 127;
  ge_cat[(size_t)b * 256 + colOff + c] = (float)h[(size_t)ind[b] * HH + c];
}

__global__ void logits_kernel(const float* __restrict__ hidden,
                              const float* __restrict__ W2,
                              const float* __restrict__ b2,
                              float* __restrict__ out) {
  int i = blockIdx.x * blockDim.x + threadIdx.x;
  if (i >= BB) return;
  const float* hr = hidden + (size_t)i * HH;
  float acc = b2[0];
#pragma unroll
  for (int j = 0; j < HH; j += 4) {
    float4 hv = *(const float4*)(hr + j);
    float4 wv = *(const float4*)(W2 + j);
    acc += hv.x * wv.x + hv.y * wv.y + hv.z * wv.z + hv.w * wv.w;
  }
  out[i] = sigm(acc);
}

__global__ void loss_kernel(const float* __restrict__ logits,
                            const int* __restrict__ labels,
                            float* __restrict__ out) {
  __shared__ float red[256];
  float acc = 0.0f;
  for (int i = threadIdx.x; i < BB; i += 256) {
    float p = logits[i];
    float y = (float)labels[i];
    acc += y * __logf(p + 1e-12f) + (1.0f - y) * __logf(1.0f - p + 1e-12f);
  }
  red[threadIdx.x] = acc;
  __syncthreads();
  for (int s = 128; s > 0; s >>= 1) {
    if ((int)threadIdx.x < s) red[threadIdx.x] += red[threadIdx.x + s];
    __syncthreads();
  }
  if (threadIdx.x == 0) out[0] = -red[0] / (float)BB;
}

// ---------------------------------------------------------------------------
// Host orchestration
// ---------------------------------------------------------------------------
extern "C" void kernel_launch(void* const* d_in, const int* in_sizes, int n_in,
                              void* d_out, int out_size, void* d_ws, size_t ws_size,
                              hipStream_t stream) {
  const int* emb_ind[2] = {(const int*)d_in[0], (const int*)d_in[1]};
  const int* adj[2]     = {(const int*)d_in[2], (const int*)d_in[3]};
  const int* prop_ind[2]= {(const int*)d_in[4], (const int*)d_in[5]};
  const int*   labels   = (const int*)d_in[6];
  const float* emb      = (const float*)d_in[7];
  const float* proj_W   = (const float*)d_in[8];
  const float* proj_b   = (const float*)d_in[9];
  const float* msg_W    = (const float*)d_in[10];
  const float* msg_b    = (const float*)d_in[11];
  const float* gru_Wih  = (const float*)d_in[12];
  const float* gru_Whh  = (const float*)d_in[13];
  const float* gru_bih  = (const float*)d_in[14];
  const float* gru_bhh  = (const float*)d_in[15];
  const float* cla_W1   = (const float*)d_in[16];
  const float* cla_b1   = (const float*)d_in[17];
  const float* cla_W2   = (const float*)d_in[18];
  const float* cla_b2   = (const float*)d_in[19];
  float* out = (float*)d_out;
  (void)in_sizes; (void)n_in; (void)out_size; (void)ws_size;

  // --- carve workspace ---
  char* ws = (char*)d_ws;
  size_t off = 0;
  auto carve = [&](size_t bytes) -> void* {
    void* p = ws + off;
    off = (off + bytes + 255) & ~(size_t)255;
    return p;
  };
  __bf16* wb_proj = (__bf16*)carve((size_t)HH * DDP * 2);
  __bf16* wb_msg  = (__bf16*)carve((size_t)LL * HH * HH * 2);
  __bf16* wb_ih   = (__bf16*)carve((size_t)LL * 3 * HH * HH * 2);
  __bf16* wb_hh   = (__bf16*)carve((size_t)LL * 3 * HH * HH * 2);
  __bf16* wb_c1   = (__bf16*)carve((size_t)HH * 2 * HH * 2);
  __bf16* hA    = (__bf16*)carve((size_t)NN * HH * 2);
  __bf16* hB    = (__bf16*)carve((size_t)NN * HH * 2);
  float*  inc   = (float*)carve((size_t)NN * HH * 4);
  float*  ge_cat= (float*)carve((size_t)BB * 2 * HH * 4);
  float*  hidden= (float*)carve((size_t)BB * HH * 4);

  // --- weights -> bf16 (pad proj K 300 -> 320) ---
  auto cvt = [&](const float* s, __bf16* d, int rows, int K, int Kpad) {
    int n = rows * Kpad;
    cvt_bf16_kernel<<<(n + 255) / 256, 256, 0, stream>>>(s, d, rows, K, Kpad);
  };
  cvt(proj_W,  wb_proj, HH,          DD, DDP);
  cvt(msg_W,   wb_msg,  LL * HH,     HH, HH);
  cvt(gru_Wih, wb_ih,   LL * 3 * HH, HH, HH);
  cvt(gru_Whh, wb_hh,   LL * 3 * HH, HH, HH);
  cvt(cla_W1,  wb_c1,   HH,      2 * HH, 2 * HH);

  for (int g = 0; g < 2; ++g) {
    __bf16* h = hA;
    __bf16* hn = hB;
    {  // projection: h = bf16( emb[emb_ind] @ proj_W^T + proj_b )
      int rowTiles = NN / 16, colTiles = HH / 16;
      int tiles = rowTiles * colTiles;
      gather_gemm_kernel<true, __bf16><<<(tiles + 7) / 8, 256, 0, stream>>>(
          emb, DD, DD, DDP / 32, emb_ind[g], wb_proj, DDP, proj_b,
          h, HH, rowTiles, colTiles, /*act=*/0);
    }
    for (int layer = 0; layer < LL; ++layer) {
      for (int t = 0; t < TT; ++t) {
        hipMemsetAsync(inc, 0, (size_t)NN * HH * 4, stream);
        int eTiles = EE / 16;
        msg_scatter_kernel<<<(eTiles + 7) / 8, 256, 0, stream>>>(
            h, adj[g], wb_msg + (size_t)layer * HH * HH, msg_b + layer * HH,
            inc, eTiles);
        int tiles = (NN / 16) * 8;
        gru_kernel<<<(tiles + 7) / 8, 256, 0, stream>>>(
            inc, h, wb_ih + (size_t)layer * 3 * HH * HH,
            wb_hh + (size_t)layer * 3 * HH * HH,
            gru_bih + layer * 3 * HH, gru_bhh + layer * 3 * HH, hn, NN / 16);
        __bf16* tmp = h; h = hn; hn = tmp;
      }
    }
    prop_gather_kernel<<<(BB * HH + 255) / 256, 256, 0, stream>>>(
        h, prop_ind[g], ge_cat, g * HH);
  }

  {  // hidden = relu(ge_cat @ cla_W1^T + b1)
    int rowTiles = BB / 16, colTiles = HH / 16;
    int tiles = rowTiles * colTiles;
    gather_gemm_kernel<false, float><<<(tiles + 7) / 8, 256, 0, stream>>>(
        ge_cat, 2 * HH, 2 * HH, (2 * HH) / 32, nullptr, wb_c1, 2 * HH, cla_b1,
        hidden, HH, rowTiles, colTiles, /*act=*/1);
  }
  logits_kernel<<<(BB + 127) / 128, 128, 0, stream>>>(hidden, cla_W2, cla_b2, out);
  loss_kernel<<<1, 256, 0, stream>>>(out, labels, out + BB);
}
